// BertSelfAttention_50869592654354
// MI455X (gfx1250) — matrix operand order repaired and checked by reading, not compiled, not run on hardware
//
#include <hip/hip_runtime.h>

// ---------------------------------------------------------------------------
// BERT self-attention fwd for MI455X (gfx1250, wave32, WMMA bf16 16x16x32)
// B=4, S=2048, H=768, NH=12, HD=64
// Pipeline: prep (f32->bf16 repack) -> fused QKV GEMM -> flash attention.
// Global->LDS staging uses gfx1250 async-to-LDS copies with double buffering.
// ---------------------------------------------------------------------------

typedef __attribute__((ext_vector_type(16))) __bf16 v16bf;
typedef __attribute__((ext_vector_type(8)))  float  v8f;

#define NB    4
#define SEQ   2048
#define HID   768
#define NHEAD 12
#define HDIM  64
#define NQKV  2304               // 3*HID concatenated Q,K,V output features
#define MROWS (NB * SEQ)         // 8192
#define LOG2E 1.44269504088896f
#define ATT_SCALE 0.125f         // 1/sqrt(64)

// -------------------- gfx1250 async global->LDS copy -----------------------

#if __has_builtin(__builtin_amdgcn_global_load_async_to_lds_b128)
#define HAVE_ASYNC 1
#else
#define HAVE_ASYNC 0
#endif

// Builtin prototype is (v4i AS1*, v4i AS3*, imm offset, imm cpol)
typedef int async_i4 __attribute__((ext_vector_type(4)));
typedef __attribute__((address_space(1))) async_i4 g_async_i4;
typedef __attribute__((address_space(3))) async_i4 l_async_i4;

__device__ __forceinline__ void g2l_b128(void* l, const void* g) {
#if HAVE_ASYNC
    // AS1 = flat 64-bit address; AS3 offset = low 32 bits of flat LDS address
    __builtin_amdgcn_global_load_async_to_lds_b128(
        (g_async_i4*)(unsigned long)g,
        (l_async_i4*)(unsigned)(unsigned long)l,
        /*offset=*/0, /*cpol=*/0);
#else
    *(uint4*)l = *(const uint4*)g;
#endif
}

__device__ __forceinline__ void wait_async0() {
#if HAVE_ASYNC
#if __has_builtin(__builtin_amdgcn_s_wait_asynccnt)
    __builtin_amdgcn_s_wait_asynccnt(0);
#else
    asm volatile("s_wait_asynccnt 0x0" ::: "memory");
#endif
#endif
}

// -------------------- fragment helpers (wave32 layouts) --------------------

union FragU { v16bf v; unsigned u[8]; };

// A-matrix 16x32 bf16 fragment gather from a row-major [16][stride] tile.
// lane&15 = M row; lane>>4 selects K half (k = 16*hi .. 16*hi+15, contiguous).
__device__ __forceinline__ v16bf load_a_frag(const __bf16* p, int stride) {
    int lane = threadIdx.x & 31;
    int m = lane & 15, hi = lane >> 4;
    FragU f;
    const unsigned* rp = (const unsigned*)(p + m * stride);
#pragma unroll
    for (int j = 0; j < 8; ++j) {
        int k = 16 * hi + 2 * j;  // 16*hi + 2*j (contiguous K per lane half)
        f.u[j] = rp[k >> 1];
    }
    return f.v;
}

// B-matrix 32x16 bf16 fragment gather from an [n][stride] tile (n-major,
// i.e. tile[n][k] = B[k][n]).  lane&15 = N col; lane>>4 selects K half.
__device__ __forceinline__ v16bf load_b_frag(const __bf16* p, int stride) {
    int lane = threadIdx.x & 31;
    int n = lane & 15, hi = lane >> 4;
    FragU f;
    const unsigned* rp = (const unsigned*)(p + n * stride);
#pragma unroll
    for (int j = 0; j < 8; ++j)
        f.u[j] = rp[(16 * hi + 2 * j) >> 1];
    return f.v;
}

// B-matrix 32x16 bf16 fragment gather from a row-major [k][stride] tile
// (tile[k][n] = B[k][n], e.g. V kept verbatim as [key][d]).
__device__ __forceinline__ v16bf load_b_frag_rowmajor(const __bf16* p, int stride) {
    int lane = threadIdx.x & 31;
    int n = lane & 15, hi = lane >> 4;
    v16bf f;
#pragma unroll
    for (int e = 0; e < 16; ++e)
        f[e] = p[(16 * hi + e) * stride + n];
    return f;
}

__device__ __forceinline__ v8f wmma_bf16(v16bf a, v16bf b, v8f c) {
    return __builtin_amdgcn_wmma_f32_16x16x32_bf16(
        /*neg_a=*/false, a, /*neg_b=*/false, b,
        /*c_mod=*/(short)0, c, /*reuse_a=*/false, /*reuse_b=*/false);
}

// -------------------- kernel 0: precision convert / repack -----------------

__global__ __launch_bounds__(256) void prep_kernel(
    const float* __restrict__ hs,
    const float* __restrict__ Wq, const float* __restrict__ Wk,
    const float* __restrict__ Wv,
    const float* __restrict__ bq, const float* __restrict__ bk,
    const float* __restrict__ bv,
    __bf16* __restrict__ Xb, __bf16* __restrict__ Wc,
    float* __restrict__ biasc) {
    const long NX = (long)MROWS * HID;
    const long NW = (long)HID * HID;
    long step = (long)gridDim.x * blockDim.x;
    long i0 = (long)blockIdx.x * blockDim.x + threadIdx.x;
    for (long i = i0; i < NX; i += step) Xb[i] = (__bf16)hs[i];
    for (long i = i0; i < NW; i += step) {
        Wc[i]          = (__bf16)Wq[i];   // rows   0..767  : Q features
        Wc[NW + i]     = (__bf16)Wk[i];   // rows 768..1535 : K features
        Wc[2 * NW + i] = (__bf16)Wv[i];   // rows 1536..2303: V features
    }
    for (long i = i0; i < HID; i += step) {
        biasc[i]           = bq[i];
        biasc[HID + i]     = bk[i];
        biasc[2 * HID + i] = bv[i];
    }
}

// -------------------- kernel 1: fused QKV GEMM (bf16 WMMA) -----------------
// C[8192,2304] = X[8192,768] * Wc^T  (+ bias), stored bf16.
// Block tile 128x128, K chunk 64, 8 waves -> each wave 32x64 (2x4 frags).
// Double-buffered LDS tiles filled by async global->LDS copies.

#define MT 128
#define NT 128
#define KT 64

__global__ __launch_bounds__(256) void qkv_gemm_kernel(
    const __bf16* __restrict__ X, const __bf16* __restrict__ W,
    const float* __restrict__ biasc, __bf16* __restrict__ QKV) {
    __shared__ __align__(16) __bf16 sA[2][MT * KT];   // [m][k] row-major
    __shared__ __align__(16) __bf16 sB[2][NT * KT];   // [n][k] row-major (= B^T)

    const int m0 = blockIdx.x * MT;
    const int n0 = blockIdx.y * NT;
    const int tid = threadIdx.x;
    const int wave = tid >> 5, lane = tid & 31;
    const int wm = wave & 3;           // 4 waves along M (32 rows each)
    const int wn = wave >> 2;          // 2 waves along N (64 cols each)

    v8f acc[2][4] = {};

    const int r = tid >> 1, hseg = tid & 1;   // 2 threads per tile row, 64B each

    auto issueAB = [&](int kc, int p) {
        const __bf16* xg = X + (long)(m0 + r) * HID + kc + hseg * 32;
        const __bf16* wg = W + (long)(n0 + r) * HID + kc + hseg * 32;
        __bf16* xl = sA[p] + r * KT + hseg * 32;
        __bf16* wl = sB[p] + r * KT + hseg * 32;
#pragma unroll
        for (int q = 0; q < 4; ++q) {
            g2l_b128(xl + q * 8, xg + q * 8);
            g2l_b128(wl + q * 8, wg + q * 8);
        }
    };

    issueAB(0, 0);
    for (int kc = 0; kc < HID; kc += KT) {
        const int p = (kc / KT) & 1;
        wait_async0();           // current tile's async copies done (this wave)
        __syncthreads();         // all waves' copies visible; prev buf free
        if (kc + KT < HID) issueAB(kc + KT, 1 - p);   // overlap with compute
#pragma unroll
        for (int ks = 0; ks < KT; ks += 32) {
            v16bf a0 = load_a_frag(sA[p] + (wm * 32 + 0)  * KT + ks, KT);
            v16bf a1 = load_a_frag(sA[p] + (wm * 32 + 16) * KT + ks, KT);
            v16bf b0 = load_b_frag(sB[p] + (wn * 64 + 0)  * KT + ks, KT);
            v16bf b1 = load_b_frag(sB[p] + (wn * 64 + 16) * KT + ks, KT);
            v16bf b2 = load_b_frag(sB[p] + (wn * 64 + 32) * KT + ks, KT);
            v16bf b3 = load_b_frag(sB[p] + (wn * 64 + 48) * KT + ks, KT);
            acc[0][0] = wmma_bf16(a0, b0, acc[0][0]);
            acc[0][1] = wmma_bf16(a0, b1, acc[0][1]);
            acc[0][2] = wmma_bf16(a0, b2, acc[0][2]);
            acc[0][3] = wmma_bf16(a0, b3, acc[0][3]);
            acc[1][0] = wmma_bf16(a1, b0, acc[1][0]);
            acc[1][1] = wmma_bf16(a1, b1, acc[1][1]);
            acc[1][2] = wmma_bf16(a1, b2, acc[1][2]);
            acc[1][3] = wmma_bf16(a1, b3, acc[1][3]);
        }
        __syncthreads();         // all waves done with buf p before refill
    }

    // epilogue: bias add, bf16 store. D-frag: row = e + 8*(lane>>4), col = lane&15
    const int nn = lane & 15, hi = lane >> 4;
#pragma unroll
    for (int i = 0; i < 2; ++i) {
#pragma unroll
        for (int j = 0; j < 4; ++j) {
            int col = n0 + wn * 64 + j * 16 + nn;
            float bias = biasc[col];
#pragma unroll
            for (int e = 0; e < 8; ++e) {
                int row = m0 + wm * 32 + i * 16 + e + 8 * hi;
                QKV[(long)row * NQKV + col] = (__bf16)(acc[i][j][e] + bias);
            }
        }
    }
}

// -------------------- kernel 2: flash attention (bf16 WMMA) ----------------
// grid: (B*NH, S/128). Block = 8 waves; each wave owns 16 query rows.
// Streams keys/values in 32-wide chunks (double-buffered async) with online
// softmax; scores never touch HBM.

#define QT 128
#define KC 32

__global__ __launch_bounds__(256) void attn_kernel(
    const __bf16* __restrict__ QKV, const float* __restrict__ mask,
    float* __restrict__ out) {
    __shared__ __align__(16) __bf16 sK[2][KC * HDIM];   // [key][d] (= scores B^T)
    __shared__ __align__(16) __bf16 sV[2][KC * HDIM];   // [key][d] verbatim
    __shared__ __align__(16) __bf16 sP[8][16 * KC];     // per-wave P tile

    const int bh = blockIdx.x;
    const int b = bh / NHEAD, h = bh % NHEAD;
    const int q0 = blockIdx.y * QT;
    const int tid = threadIdx.x, wave = tid >> 5, lane = tid & 31;
    const int qr = q0 + wave * 16;

    const __bf16* Qbase = QKV + (long)(b * SEQ) * NQKV + h * HDIM;
    const __bf16* Kbase = Qbase + HID;
    const __bf16* Vbase = Qbase + 2 * HID;
    const float* mrow = mask + (long)b * SEQ;

    // Q fragments gathered straight from global (A layout, row stride NQKV)
    v16bf qa0 = load_a_frag(Qbase + (long)qr * NQKV, NQKV);
    v16bf qa1 = load_a_frag(Qbase + (long)qr * NQKV + 32, NQKV);

    float mi[8], li[8];
#pragma unroll
    for (int e = 0; e < 8; ++e) { mi[e] = -1.0e30f; li[e] = 0.0f; }
    v8f ctx[4] = {};

    const float kScale = ATT_SCALE * LOG2E;   // fold log2(e) -> use exp2
    const int krow = tid >> 3, kseg = tid & 7;
    const int nn = lane & 15, hi = lane >> 4;

    auto issueKV = [&](int k0, int p) {
        const __bf16* kg = Kbase + (long)(k0 + krow) * NQKV + kseg * 8;
        const __bf16* vg = Vbase + (long)(k0 + krow) * NQKV + kseg * 8;
        g2l_b128(sK[p] + krow * HDIM + kseg * 8, kg);
        g2l_b128(sV[p] + krow * HDIM + kseg * 8, vg);
    };

    issueKV(0, 0);
    for (int k0 = 0; k0 < SEQ; k0 += KC) {
        const int p = (k0 / KC) & 1;
        wait_async0();
        __syncthreads();
        if (k0 + KC < SEQ) issueKV(k0 + KC, 1 - p);   // overlap with compute

        // scores: two 16x16 tiles (keys 0..15, 16..31), K-dim = HDIM in 2 steps
        v8f s0 = {}, s1 = {};
        s0 = wmma_bf16(qa0, load_b_frag(sK[p] + 0 * HDIM + 0,  HDIM), s0);
        s0 = wmma_bf16(qa1, load_b_frag(sK[p] + 0 * HDIM + 32, HDIM), s0);
        s1 = wmma_bf16(qa0, load_b_frag(sK[p] + 16 * HDIM + 0,  HDIM), s1);
        s1 = wmma_bf16(qa1, load_b_frag(sK[p] + 16 * HDIM + 32, HDIM), s1);

        const float mv0 = mrow[k0 + nn] * LOG2E;
        const float mv1 = mrow[k0 + 16 + nn] * LOG2E;
        __bf16* pw = sP[wave];

        // online softmax (log2 domain); rows live in element e, cols in lanes
#pragma unroll
        for (int e = 0; e < 8; ++e) {
            float a = s0[e] * kScale + mv0;
            float c = s1[e] * kScale + mv1;
            float m = fmaxf(a, c);
#pragma unroll
            for (int d = 1; d < 16; d <<= 1) m = fmaxf(m, __shfl_xor(m, d, 32));
            float nm = fmaxf(mi[e], m);
            float al = exp2f(mi[e] - nm);
            mi[e] = nm;
            float p0 = exp2f(a - nm);
            float p1 = exp2f(c - nm);
            float rsum = p0 + p1;
#pragma unroll
            for (int d = 1; d < 16; d <<= 1) rsum += __shfl_xor(rsum, d, 32);
            li[e] = li[e] * al + rsum;
#pragma unroll
            for (int j = 0; j < 4; ++j) ctx[j][e] *= al;
            int prow = e + 8 * hi;
            pw[prow * KC + nn]      = (__bf16)p0;   // D-frag -> row-major P tile
            pw[prow * KC + 16 + nn] = (__bf16)p1;
        }

        // intra-wave LDS ordering: DS ops are in-order per wave; block compiler
        // reordering and drain the DS counter before re-reading as A fragment.
        asm volatile("s_wait_dscnt 0" ::: "memory");

        v16bf pa = load_a_frag(pw, KC);   // P as 16x32 A fragment
#pragma unroll
        for (int j = 0; j < 4; ++j)
            ctx[j] = wmma_bf16(pa, load_b_frag_rowmajor(sV[p] + j * 16, HDIM),
                               ctx[j]);
        __syncthreads();   // all waves done with buf p before it is refilled
    }

    // normalize and store f32 output [B,S,H]
    float inv[8];
#pragma unroll
    for (int e = 0; e < 8; ++e) inv[e] = 1.0f / li[e];
#pragma unroll
    for (int j = 0; j < 4; ++j) {
#pragma unroll
        for (int e = 0; e < 8; ++e) {
            int row = qr + e + 8 * hi;
            out[((long)(b * SEQ + row)) * HID + h * HDIM + j * 16 + nn] =
                ctx[j][e] * inv[e];
        }
    }
}

// -------------------- host-side launcher -----------------------------------

extern "C" void kernel_launch(void* const* d_in, const int* in_sizes, int n_in,
                              void* d_out, int out_size, void* d_ws, size_t ws_size,
                              hipStream_t stream) {
    const float* hs   = (const float*)d_in[0];
    const float* mask = (const float*)d_in[1];
    const float* Wq   = (const float*)d_in[2];
    const float* bq   = (const float*)d_in[3];
    const float* Wk   = (const float*)d_in[4];
    const float* bk   = (const float*)d_in[5];
    const float* Wv   = (const float*)d_in[6];
    const float* bv   = (const float*)d_in[7];
    float* out = (float*)d_out;

    // workspace layout (bytes)
    char* ws = (char*)d_ws;
    __bf16* Xb    = (__bf16*)ws;                         // 8192*768*2  = 12,582,912
    __bf16* Wc    = (__bf16*)(ws + 12582912);            // 2304*768*2  =  3,538,944
    float*  biasc = (float*)(ws + 16121856);             // 2304*4      (padded 16K)
    __bf16* QKV   = (__bf16*)(ws + 16138240);            // 8192*2304*2 = 37,748,736

    prep_kernel<<<dim3(512), 256, 0, stream>>>(hs, Wq, Wk, Wv, bq, bk, bv,
                                               Xb, Wc, biasc);
    qkv_gemm_kernel<<<dim3(MROWS / MT, NQKV / NT), 256, 0, stream>>>(Xb, Wc,
                                                                     biasc, QKV);
    attn_kernel<<<dim3(NB * NHEAD, SEQ / QT), 256, 0, stream>>>(QKV, mask, out);
}
